// UCBAttention_5506148073506
// MI455X (gfx1250) — matrix operand order repaired and checked by reading, not compiled, not run on hardware
//
#include <hip/hip_runtime.h>
#include <hip/hip_bf16.h>

#define B_      16
#define N_      577
#define DIM_    768
#define H_      12
#define DH_     64
#define NPAD    640
#define ROWS    (B_ * N_)        // 9232
#define OUTCOLS (3 * DIM_)       // 2304
#define KKEEP   288

typedef __attribute__((ext_vector_type(16))) _Float16 v16h;
typedef __attribute__((ext_vector_type(8)))  float    v8f;

union FragU { v16h v; float4 f4[2]; };

__device__ inline v8f vzero() { v8f z; for (int i = 0; i < 8; i++) z[i] = 0.f; return z; }

// ---- CDNA5 async global->LDS path (ASYNCcnt), guarded so the file compiles either way ----
#if defined(__HIP_DEVICE_COMPILE__)
#if __has_builtin(__builtin_amdgcn_global_load_async_to_lds_b128) && __has_builtin(__builtin_amdgcn_s_wait_asynccnt)
#define HAVE_ASYNC 1
#endif
#endif

#if HAVE_ASYNC
typedef int i4_ __attribute__((vector_size(16)));                 // matches builtin's int4 param
typedef __attribute__((address_space(1))) i4_ GI4;                // global (prints as __device__)
typedef __attribute__((address_space(3))) i4_ LI4;                // LDS
#endif

__device__ inline void copy16_async(const _Float16* g, _Float16* l) {
#if HAVE_ASYNC
  __builtin_amdgcn_global_load_async_to_lds_b128((GI4*)g, (LI4*)l, 0, 0);
#else
  *reinterpret_cast<float4*>(l) = *reinterpret_cast<const float4*>(g);
#endif
}
__device__ inline void wait_async() {
#if HAVE_ASYNC
  __builtin_amdgcn_s_wait_asynccnt(0);
#endif
}

// A-fragment 16x32 f16 (wave32): lane = 16*hi + row; chunks at half-offsets hi*8 and 16+hi*8
__device__ inline v16h load_a(const _Float16* base, int ld, int lane) {
  FragU f;
  int row = lane & 15;
  int hi  = (lane >> 4) & 1;
  const _Float16* p = base + row * ld + hi * 8;
  f.f4[0] = *reinterpret_cast<const float4*>(p);
  f.f4[1] = *reinterpret_cast<const float4*>(p + 16);
  return f.v;
}
// B-fragment 32x16 f16: lane = 16*hi + col; k = hi*8+s (s<8), 16+hi*8+(s-8) (s>=8)
__device__ inline v16h load_b(const _Float16* base, int ld, int lane) {
  v16h b;
  int col = lane & 15;
  int hi  = (lane >> 4) & 1;
  for (int s = 0; s < 16; s++) {
    int k = (s < 8) ? (hi * 8 + s) : (16 + hi * 8 + (s - 8));
    b[s] = base[k * ld + col];
  }
  return b;
}
__device__ inline v8f wmma16(v16h a, v16h b, v8f c) {
  return __builtin_amdgcn_wmma_f32_16x16x32_f16(false, a, false, b, (short)0, c, false, false);
}

// ---------------- conversion kernels ----------------
__global__ void k_cvt_f16(const float* __restrict__ in, _Float16* __restrict__ outp, int n) {
  int i = blockIdx.x * 256 + threadIdx.x;
  if (i < n) outp[i] = (_Float16)in[i];
}
// in: (O x K) row-major -> out: (K x O) f16  (out[k*O+o] = in[o*K+k])
__global__ void k_cvt_transpose(const float* __restrict__ in, _Float16* __restrict__ outp, int O, int K) {
  int i = blockIdx.x * 256 + threadIdx.x;
  if (i < O * K) { int o = i / K, k = i - o * K; outp[k * O + o] = (_Float16)in[i]; }
}

// ---------------- shared GEMM mainloop: 64x64 tile, K-step 32, double-buffered async LDS ----
__device__ inline void gemm_tiles(const _Float16* __restrict__ A,
                                  const _Float16* __restrict__ Bm, int ldb,
                                  _Float16* As, _Float16* Bs,
                                  int M0, int N0, int t, v8f c[2][2]) {
  int lane = t & 31, wave = t >> 5;
  int wm = wave >> 1, wn = wave & 1;
  auto issue = [&](int k0, int bi) {
    for (int cc2 = t; cc2 < 256; cc2 += 128) {           // A tile 64x32
      int row = cc2 >> 2, cc = cc2 & 3;
      int gr = M0 + row;
      if (gr >= ROWS) gr = ROWS - 1;                     // clamp: surplus rows discarded at store
      copy16_async(A + (size_t)gr * DIM_ + k0 + cc * 8, As + bi * 2048 + row * 32 + cc * 8);
    }
    for (int cc2 = t; cc2 < 256; cc2 += 128) {           // B tile 32x64
      int row = cc2 >> 3, cc = cc2 & 7;
      copy16_async(Bm + (size_t)(k0 + row) * ldb + N0 + cc * 8, Bs + bi * 2048 + row * 64 + cc * 8);
    }
  };
  issue(0, 0);
  wait_async();
  __syncthreads();
  for (int ks = 0; ks < DIM_ / 32; ks++) {
    int cur = ks & 1;
    if (ks + 1 < DIM_ / 32) issue((ks + 1) * 32, cur ^ 1);  // prefetch next K-tile
    const _Float16* as = As + cur * 2048;
    const _Float16* bs = Bs + cur * 2048;
    v16h a0 = load_a(as + (wm * 32) * 32, 32, lane);
    v16h a1 = load_a(as + (wm * 32 + 16) * 32, 32, lane);
    v16h b0 = load_b(bs + wn * 32, 64, lane);
    v16h b1 = load_b(bs + wn * 32 + 16, 64, lane);
    c[0][0] = wmma16(a0, b0, c[0][0]);
    c[0][1] = wmma16(a0, b1, c[0][1]);
    c[1][0] = wmma16(a1, b0, c[1][0]);
    c[1][1] = wmma16(a1, b1, c[1][1]);
    __syncthreads();          // all waves done reading cur + issued next
    wait_async();             // own async copies for next buffer landed
    __syncthreads();          // everyone's landed
  }
}

// ---------------- QKV projection GEMM: scatter to q(*scale), kt (transposed), v ----------------
__global__ __launch_bounds__(128) void k_qkv_gemm(const _Float16* __restrict__ A,
                                                  const _Float16* __restrict__ Bm,
                                                  _Float16* __restrict__ q,
                                                  _Float16* __restrict__ kt,
                                                  _Float16* __restrict__ v) {
  __shared__ _Float16 As[2 * 2048];
  __shared__ _Float16 Bs[2 * 2048];
  int t = threadIdx.x, lane = t & 31, wave = t >> 5;
  int wm = wave >> 1, wn = wave & 1;
  int M0 = blockIdx.y * 64, N0 = blockIdx.x * 64;
  v8f c[2][2];
  c[0][0] = vzero(); c[0][1] = vzero(); c[1][0] = vzero(); c[1][1] = vzero();
  gemm_tiles(A, Bm, OUTCOLS, As, Bs, M0, N0, t, c);
  int hi = lane >> 4, lo = lane & 15;
  for (int rg = 0; rg < 2; rg++)
    for (int cg = 0; cg < 2; cg++)
      for (int r = 0; r < 8; r++) {
        int m = M0 + wm * 32 + rg * 16 + r + 8 * hi;
        int n = N0 + wn * 32 + cg * 16 + lo;
        if (m >= ROWS) continue;
        float val = c[rg][cg][r];
        int b = m / N_, nn = m - b * N_;
        int which = n / DIM_, rem = n - which * DIM_;
        int h = rem >> 6, d = rem & 63;
        size_t bh = (size_t)(b * H_ + h);
        if (which == 0)      q[(bh * NPAD + nn) * DH_ + d] = (_Float16)(val * 0.125f); // Dh^-0.5
        else if (which == 1) kt[(bh * DH_ + d) * NPAD + nn] = (_Float16)val;           // K transposed
        else                 v[(bh * NPAD + nn) * DH_ + d] = (_Float16)val;
      }
}

// ---------------- attention pass 1: row inverse-sums + column sums (max-free softmax) ----------
__global__ __launch_bounds__(256) void k_attn_stats(const _Float16* __restrict__ q,
                                                    const _Float16* __restrict__ kt,
                                                    float* __restrict__ rowinv,
                                                    float* __restrict__ colsum) {
  int bh = blockIdx.x;
  int lane = threadIdx.x & 31, wave = threadIdx.x >> 5;
  int q0 = blockIdx.y * 128 + wave * 16;
  if (q0 >= N_) return;
  const _Float16* Qb = q + (size_t)bh * NPAD * DH_;
  const _Float16* Kb = kt + (size_t)bh * DH_ * NPAD;
  int hi = lane >> 4, lo = lane & 15;
  v16h a0 = load_a(Qb + q0 * DH_, DH_, lane);        // d 0..31
  v16h a1 = load_a(Qb + q0 * DH_ + 32, DH_, lane);   // d 32..63
  float acc[8], rv[8];
  for (int r = 0; r < 8; r++) {
    acc[r] = 0.f;
    rv[r] = ((q0 + r + 8 * hi) < N_) ? 1.f : 0.f;
  }
  for (int tt = 0; tt < 37; tt++) {                  // sweep 1: per-lane partial row sums
    int n0 = tt * 16;
    v16h b0 = load_b(Kb + n0, NPAD, lane);
    v16h b1 = load_b(Kb + 32 * NPAD + n0, NPAD, lane);
    v8f s = vzero();
    s = wmma16(a0, b0, s);
    s = wmma16(a1, b1, s);
    float cm = ((n0 + lo) < N_) ? 1.f : 0.f;
    for (int r = 0; r < 8; r++) acc[r] += cm * __expf(s[r]);
  }
  float invl[8];
  for (int r = 0; r < 8; r++) {                      // one reduction per row (not per tile)
    float l = acc[r];
    for (int msk = 1; msk < 16; msk <<= 1) l += __shfl_xor(l, msk, 32);
    invl[r] = rv[r] / fmaxf(l, 1e-30f);
  }
  if (lo == 0)
    for (int r = 0; r < 8; r++) {
      int row = q0 + r + 8 * hi;
      if (row < N_) rowinv[(size_t)bh * NPAD + row] = invl[r];
    }
  for (int tt = 0; tt < 37; tt++) {                  // sweep 2: column sums of probs
    int n0 = tt * 16;
    v16h b0 = load_b(Kb + n0, NPAD, lane);
    v16h b1 = load_b(Kb + 32 * NPAD + n0, NPAD, lane);
    v8f s = vzero();
    s = wmma16(a0, b0, s);
    s = wmma16(a1, b1, s);
    float part = 0.f;
    for (int r = 0; r < 8; r++) part += __expf(s[r]) * invl[r];  // invl=0 kills pad rows
    part += __shfl_xor(part, 16, 32);
    if (lane < 16 && (n0 + lo) < N_) atomicAdd(&colsum[(size_t)bh * NPAD + n0 + lo], part);
  }
}

// ---------------- UCB scoring + exact top-k ----------------
__global__ __launch_bounds__(256) void k_ucb_topk(const float* __restrict__ colsum,
                                                  const float* __restrict__ counts,
                                                  const int* __restrict__ counter,
                                                  const int* __restrict__ ucb_en,
                                                  float* __restrict__ keep,
                                                  float* __restrict__ delta) {
  int b = blockIdx.x, t = threadIdx.x;
  __shared__ float gs[576];
  bool active = (ucb_en[0] != 0) && (counter[0] >= 50);
  if (!active) {
    for (int n = t; n < N_; n += 256) keep[b * NPAD + n] = 1.f;
    return;
  }
  float logc = __logf((float)counter[0] + 1.0f);
  for (int n = t; n < 576; n += 256) {
    float acc = 0.f;
    for (int h = 0; h < H_; h++) {
      acc += colsum[((size_t)(b * H_ + h)) * NPAD + n + 1] * (1.f / (float)N_)
           + sqrtf(logc / (counts[h * N_ + n + 1] + 1e-6f));
    }
    gs[n] = acc * (1.f / (float)H_);
  }
  __syncthreads();
  for (int n = t; n < 576; n += 256) {
    float vv = gs[n];
    int rank = 0;
    for (int m2 = 0; m2 < 576; m2++) {
      float u = gs[m2];
      rank += (u > vv) || (u == vv && m2 < n);       // top_k tie-break: first index wins
    }
    if (rank < KKEEP) {
      keep[b * NPAD + n + 1] = 1.f;
      for (int h = 0; h < H_; h++) atomicAdd(&delta[h * N_ + n + 1], 1.0f / (float)B_);
    }
  }
  if (t == 0) keep[b * NPAD] = 1.f;                  // CLS always kept
}

// ---------------- attention pass 2: masked, renormalized P @ V ----------------
__global__ __launch_bounds__(256) void k_attn_out(const _Float16* __restrict__ q,
                                                  const _Float16* __restrict__ kt,
                                                  const _Float16* __restrict__ v,
                                                  const float* __restrict__ rowinv,
                                                  const float* __restrict__ keep,
                                                  _Float16* __restrict__ attn) {
  __shared__ _Float16 Pt[8 * 16 * 32];               // per-wave 16x32 P tile (f16)
  int bh = blockIdx.x;
  int b = bh / H_, h = bh - b * H_;
  int lane = threadIdx.x & 31, wave = threadIdx.x >> 5;
  int q0 = blockIdx.y * 128 + wave * 16;
  if (q0 >= N_) return;
  _Float16* pt = &Pt[wave * 512];
  const _Float16* Qb = q + (size_t)bh * NPAD * DH_;
  const _Float16* Kb = kt + (size_t)bh * DH_ * NPAD;
  const _Float16* Vb = v + (size_t)bh * NPAD * DH_;
  int hi = lane >> 4, lo = lane & 15;
  v16h a0 = load_a(Qb + q0 * DH_, DH_, lane);
  v16h a1 = load_a(Qb + q0 * DH_ + 32, DH_, lane);
  float il[8], kr[8], ns[8];
  for (int r = 0; r < 8; r++) {
    int row = q0 + r + 8 * hi;
    il[r] = rowinv[(size_t)bh * NPAD + row];         // 0 for pad/invalid rows
    kr[r] = keep[b * NPAD + row];
    ns[r] = 0.f;
  }
  v8f o[4];
  for (int dt = 0; dt < 4; dt++) o[dt] = vzero();
  for (int kc = 0; kc < 19; kc++) {                  // 19 x 32 keys (padded to 608 < NPAD)
    for (int sub = 0; sub < 2; sub++) {
      int n0 = kc * 32 + sub * 16;
      v16h b0 = load_b(Kb + n0, NPAD, lane);
      v16h b1 = load_b(Kb + 32 * NPAD + n0, NPAD, lane);
      v8f s = vzero();
      s = wmma16(a0, b0, s);
      s = wmma16(a1, b1, s);
      int ncol = n0 + lo;
      bool vc = ncol < N_;
      float kc2 = keep[b * NPAD + ncol];
      for (int r = 0; r < 8; r++) {
        float p = vc ? __expf(s[r]) * il[r] : 0.f;
        p *= ((kr[r] + kc2) > 0.f) ? 1.f : 0.f;      // mask = max(keep_i, keep_j)
        ns[r] += p;
        pt[(r + 8 * hi) * 32 + sub * 16 + lo] = (_Float16)p;
      }
    }
    asm volatile("s_wait_dscnt 0x0" ::: "memory");   // LDS P-tile stores -> A-frag loads
    v16h pa = load_a(pt, 32, lane);
    for (int dt = 0; dt < 4; dt++) {
      v16h vb = load_b(Vb + (size_t)kc * 32 * DH_ + dt * 16, DH_, lane);
      o[dt] = wmma16(pa, vb, o[dt]);
    }
    asm volatile("s_wait_dscnt 0x0" ::: "memory");   // keep next-iter stores behind loads
  }
  for (int r = 0; r < 8; r++) {
    float t2 = ns[r];
    for (int msk = 1; msk < 16; msk <<= 1) t2 += __shfl_xor(t2, msk, 32);
    float inv = 1.f / (t2 + 1e-8f);                  // renormalize masked probs
    int row = q0 + r + 8 * hi;
    if (row < N_) {
      for (int dt = 0; dt < 4; dt++)
        attn[((size_t)(b * N_ + row)) * DIM_ + h * DH_ + dt * 16 + lo] = (_Float16)(o[dt][r] * inv);
    }
  }
}

// ---------------- output projection GEMM (+bias) ----------------
__global__ __launch_bounds__(128) void k_proj_gemm(const _Float16* __restrict__ A,
                                                   const _Float16* __restrict__ Bm,
                                                   const float* __restrict__ bias,
                                                   float* __restrict__ outp) {
  __shared__ _Float16 As[2 * 2048];
  __shared__ _Float16 Bs[2 * 2048];
  int t = threadIdx.x, lane = t & 31, wave = t >> 5;
  int wm = wave >> 1, wn = wave & 1;
  int M0 = blockIdx.y * 64, N0 = blockIdx.x * 64;
  v8f c[2][2];
  c[0][0] = vzero(); c[0][1] = vzero(); c[1][0] = vzero(); c[1][1] = vzero();
  gemm_tiles(A, Bm, DIM_, As, Bs, M0, N0, t, c);
  int hi = lane >> 4, lo = lane & 15;
  for (int rg = 0; rg < 2; rg++)
    for (int cg = 0; cg < 2; cg++)
      for (int r = 0; r < 8; r++) {
        int m = M0 + wm * 32 + rg * 16 + r + 8 * hi;
        int n = N0 + wn * 32 + cg * 16 + lo;
        if (m < ROWS) outp[(size_t)m * DIM_ + n] = c[rg][cg][r] + bias[n];
      }
}

extern "C" void kernel_launch(void* const* d_in, const int* in_sizes, int n_in,
                              void* d_out, int out_size, void* d_ws, size_t ws_size,
                              hipStream_t stream) {
  (void)in_sizes; (void)n_in; (void)out_size; (void)ws_size;
  const float* x      = (const float*)d_in[0];
  const float* counts = (const float*)d_in[1];
  const float* wqkv   = (const float*)d_in[2];
  const float* wproj  = (const float*)d_in[3];
  const float* bproj  = (const float*)d_in[4];
  const int*   counter = (const int*)d_in[5];
  const int*   ucb_en  = (const int*)d_in[6];
  float* outp = (float*)d_out;

  char* ws = (char*)d_ws;
  size_t off = 0;
  auto alloc = [&](size_t bytes) -> char* {
    char* p = ws + off;
    off += (bytes + 255) & ~(size_t)255;
    return p;
  };
  _Float16* xh     = (_Float16*)alloc((size_t)ROWS * DIM_ * 2);
  _Float16* wqkvT  = (_Float16*)alloc((size_t)DIM_ * OUTCOLS * 2);
  _Float16* wprojT = (_Float16*)alloc((size_t)DIM_ * DIM_ * 2);
  char* zbegin = ws + off;
  _Float16* qb  = (_Float16*)alloc((size_t)B_ * H_ * NPAD * DH_ * 2);
  _Float16* ktb = (_Float16*)alloc((size_t)B_ * H_ * NPAD * DH_ * 2);
  _Float16* vb  = (_Float16*)alloc((size_t)B_ * H_ * NPAD * DH_ * 2);
  float* rowinv = (float*)alloc((size_t)B_ * H_ * NPAD * 4);
  float* colsum = (float*)alloc((size_t)B_ * H_ * NPAD * 4);
  float* keep   = (float*)alloc((size_t)B_ * NPAD * 4);
  char* zend = ws + off;
  _Float16* attn = (_Float16*)alloc((size_t)ROWS * DIM_ * 2);

  float* delta = outp + (size_t)ROWS * DIM_;

  // zero pads / accumulators / keep mask / score_delta region
  (void)hipMemsetAsync(zbegin, 0, (size_t)(zend - zbegin), stream);
  (void)hipMemsetAsync(delta, 0, (size_t)H_ * N_ * 4, stream);

  int nx = ROWS * DIM_;
  k_cvt_f16<<<(nx + 255) / 256, 256, 0, stream>>>(x, xh, nx);
  k_cvt_transpose<<<((OUTCOLS * DIM_) + 255) / 256, 256, 0, stream>>>(wqkv, wqkvT, OUTCOLS, DIM_);
  k_cvt_transpose<<<((DIM_ * DIM_) + 255) / 256, 256, 0, stream>>>(wproj, wprojT, DIM_, DIM_);

  k_qkv_gemm<<<dim3(OUTCOLS / 64, (ROWS + 63) / 64), 128, 0, stream>>>(xh, wqkvT, qb, ktb, vb);
  k_attn_stats<<<dim3(B_ * H_, 5), 256, 0, stream>>>(qb, ktb, rowinv, colsum);
  k_ucb_topk<<<B_, 256, 0, stream>>>(colsum, counts, counter, ucb_en, keep, delta);
  k_attn_out<<<dim3(B_ * H_, 5), 256, 0, stream>>>(qb, ktb, vb, rowinv, keep, attn);
  k_proj_gemm<<<dim3(DIM_ / 64, (ROWS + 63) / 64), 128, 0, stream>>>(attn, wprojT, bproj, outp);
}
